// GPT_22917945491492
// MI455X (gfx1250) — matrix operand order repaired and checked by reading, not compiled, not run on hardware
//
#include <hip/hip_runtime.h>
#include <hip/hip_bf16.h>
#include <math.h>

// ---------------------------------------------------------------------------
// GPT block forward for MI455X (gfx1250, wave32, WMMA).
// All GEMMs + both attention matmuls run on V_WMMA_F32_16X16X32_F16.
// ---------------------------------------------------------------------------

typedef __attribute__((ext_vector_type(16))) _Float16 v16h;
typedef __attribute__((ext_vector_type(8)))  _Float16 v8h;
typedef __attribute__((ext_vector_type(8)))  float    v8f;

#define Bn 4
#define Tn 2048
#define Cn 1024
#define Hn 16
#define HSn 64
#define Mrows (Bn * Tn)   // 8192

// ---------------- WMMA helpers (CDNA5 fragment layouts) --------------------

__device__ __forceinline__ v8f wmma16(v16h a, v16h b, v8f c) {
  return __builtin_amdgcn_wmma_f32_16x16x32_f16(
      /*neg_a=*/false, a, /*neg_b=*/false, b,
      /*c_mod=*/(short)0, c, /*reuse_a=*/false, /*reuse_b=*/false);
}

// A-matrix 16x32 f16 fragment. Row = m0 + lane%16.
// Element e = K[k0 + hf*16 + e] (16 contiguous halfs).
__device__ __forceinline__ v16h load_a_frag(const _Float16* __restrict__ A,
                                            int ldk, int m0, int k0,
                                            int lm, int hf) {
  const _Float16* p = A + (size_t)(m0 + lm) * ldk + k0 + hf * 16;
  v8h lo = *(const v8h*)p;
  v8h hi = *(const v8h*)(p + 8);
  v16h r;
#pragma unroll
  for (int i = 0; i < 8; ++i) { r[i] = lo[i]; r[i + 8] = hi[i]; }
  return r;
}

// B-matrix 32x16 f16 fragment from an N-major (N x K) array.
// Col = n0 + lane%16; element e = K[k0 + hf*16 + e] (16 contiguous halfs).
__device__ __forceinline__ v16h load_b_frag(const _Float16* __restrict__ Bt,
                                            int ldk, int n0, int k0,
                                            int lm, int hf) {
  return *(const v16h*)(Bt + (size_t)(n0 + lm) * ldk + k0 + hf * 16);
}

// 16-lane (half-wave) row reductions for the C-layout (N striped over lanes).
__device__ __forceinline__ float rmax16(float v) {
  v = fmaxf(v, __shfl_xor(v, 1, 32));
  v = fmaxf(v, __shfl_xor(v, 2, 32));
  v = fmaxf(v, __shfl_xor(v, 4, 32));
  v = fmaxf(v, __shfl_xor(v, 8, 32));
  return v;
}
__device__ __forceinline__ float rsum16(float v) {
  v += __shfl_xor(v, 1, 32);
  v += __shfl_xor(v, 2, 32);
  v += __shfl_xor(v, 4, 32);
  v += __shfl_xor(v, 8, 32);
  return v;
}

// ---------------- LayerNorm (f32 in -> f16 out) ----------------------------

__global__ __launch_bounds__(256) void ln_kernel(const float* __restrict__ x,
                                                 const float* __restrict__ w,
                                                 const float* __restrict__ b,
                                                 _Float16* __restrict__ out) {
  __shared__ float red[256];
  const int row = blockIdx.x;
  const int tid = threadIdx.x;
  const float* xr = x + (size_t)row * Cn;

  float s = 0.f;
  for (int c = tid; c < Cn; c += 256) s += xr[c];
  red[tid] = s; __syncthreads();
  for (int o = 128; o > 0; o >>= 1) {
    if (tid < o) red[tid] += red[tid + o];
    __syncthreads();
  }
  const float mean = red[0] * (1.0f / Cn);
  __syncthreads();

  float v = 0.f;
  for (int c = tid; c < Cn; c += 256) { float d = xr[c] - mean; v += d * d; }
  red[tid] = v; __syncthreads();
  for (int o = 128; o > 0; o >>= 1) {
    if (tid < o) red[tid] += red[tid + o];
    __syncthreads();
  }
  const float inv = rsqrtf(red[0] * (1.0f / Cn) + 1e-5f);

  for (int c = tid; c < Cn; c += 256)
    out[(size_t)row * Cn + c] = (_Float16)((xr[c] - mean) * inv * w[c] + b[c]);
}

// ---------------- Weight transpose + f16 convert (K x N -> N x K) ----------

__global__ __launch_bounds__(1024) void wtrans_kernel(const float* __restrict__ W,
                                                      _Float16* __restrict__ Wt,
                                                      int K, int N) {
  __shared__ float tile[32][33];
  int nx = blockIdx.x * 32 + threadIdx.x;  // along N
  int ky = blockIdx.y * 32 + threadIdx.y;  // along K
  if (nx < N && ky < K) tile[threadIdx.y][threadIdx.x] = W[(size_t)ky * N + nx];
  __syncthreads();
  int ko = blockIdx.y * 32 + threadIdx.x;  // along K (out inner)
  int no = blockIdx.x * 32 + threadIdx.y;  // along N (out outer)
  if (ko < K && no < N) Wt[(size_t)no * K + ko] = (_Float16)tile[threadIdx.x][threadIdx.y];
}

// ---------------- WMMA GEMM: D = A(MxK,f16) * Bt^T(NxK,f16) + bias ---------
// mode 0: out16 = D            (f16)
// mode 1: out16 = gelu(D)      (f16, exact erf gelu)
// mode 2: out32 = resid + D    (f32)

__global__ __launch_bounds__(256) void gemm_kernel(
    const _Float16* __restrict__ A, const _Float16* __restrict__ Bt,
    const float* __restrict__ bias, const float* __restrict__ resid,
    _Float16* __restrict__ out16, float* __restrict__ out32,
    int M, int N, int K, int mode) {
  const int tid  = threadIdx.x;
  const int wave = tid >> 5;
  const int lane = tid & 31;
  const int lm   = lane & 15;
  const int hf   = lane >> 4;
  const int m0   = blockIdx.x * 256 + wave * 32;  // wave tile: 32M x 64N
  const int n0   = blockIdx.y * 64;

  v8f acc[2][4] = {};
  for (int k0 = 0; k0 < K; k0 += 32) {
    v16h a0 = load_a_frag(A, K, m0,      k0, lm, hf);
    v16h a1 = load_a_frag(A, K, m0 + 16, k0, lm, hf);
#pragma unroll
    for (int ni = 0; ni < 4; ++ni) {
      v16h bf = load_b_frag(Bt, K, n0 + ni * 16, k0, lm, hf);
      acc[0][ni] = wmma16(a0, bf, acc[0][ni]);
      acc[1][ni] = wmma16(a1, bf, acc[1][ni]);
    }
  }

#pragma unroll
  for (int mi = 0; mi < 2; ++mi)
#pragma unroll
    for (int ni = 0; ni < 4; ++ni)
#pragma unroll
      for (int r = 0; r < 8; ++r) {
        const int gm = m0 + mi * 16 + r + hf * 8;
        const int gn = n0 + ni * 16 + lm;
        float v = acc[mi][ni][r] + bias[gn];
        if (mode == 1) v = 0.5f * v * (1.0f + erff(v * 0.70710678118654752f));
        if (mode == 2)
          out32[(size_t)gm * N + gn] = v + resid[(size_t)gm * N + gn];
        else
          out16[(size_t)gm * N + gn] = (_Float16)v;
      }
}

// ---------------- QKV reshape + absmax for int8 KV cache -------------------
// qkv(B,T,3C) f16 -> q(B,H,T,HS), k(B,H,T,HS), vT(B,H,HS,T) all f16.

__global__ __launch_bounds__(256) void reshape_qkv_kernel(
    const _Float16* __restrict__ qkv, _Float16* __restrict__ q,
    _Float16* __restrict__ k, _Float16* __restrict__ vt,
    unsigned int* __restrict__ absmax) {
  __shared__ float redk[256], redv[256];
  const int tid = threadIdx.x;
  const int idx = blockIdx.x * 256 + tid;     // over B*T*C
  const int c  = idx & (Cn - 1);
  const int t  = (idx >> 10) & (Tn - 1);
  const int b  = idx >> 21;
  const int h  = c >> 6, hs = c & 63;
  const size_t base = (size_t)(b * Tn + t) * (3 * Cn);
  const _Float16 qv = qkv[base + c];
  const _Float16 kv = qkv[base + Cn + c];
  const _Float16 vv = qkv[base + 2 * Cn + c];
  const size_t bh = (size_t)(b * Hn + h);
  q[(bh * Tn + t) * HSn + hs]  = qv;
  k[(bh * Tn + t) * HSn + hs]  = kv;
  vt[(bh * HSn + hs) * Tn + t] = vv;

  redk[tid] = fabsf((float)kv);
  redv[tid] = fabsf((float)vv);
  __syncthreads();
  for (int o = 128; o > 0; o >>= 1) {
    if (tid < o) {
      redk[tid] = fmaxf(redk[tid], redk[tid + o]);
      redv[tid] = fmaxf(redv[tid], redv[tid + o]);
    }
    __syncthreads();
  }
  if (tid == 0) {
    atomicMax(absmax + 0, __float_as_uint(redk[0]));  // values are >= 0
    atomicMax(absmax + 1, __float_as_uint(redv[0]));
  }
}

__global__ __launch_bounds__(256) void quantize_kernel(
    const _Float16* __restrict__ k, const _Float16* __restrict__ vt,
    const unsigned int* __restrict__ absmax,
    float* __restrict__ kq, float* __restrict__ ks,
    float* __restrict__ vq, float* __restrict__ vs) {
  const int idx = blockIdx.x * 256 + threadIdx.x;  // over B*H*T*HS
  const float kmax = __uint_as_float(absmax[0]);
  const float vmax = __uint_as_float(absmax[1]);
  const float kscale = kmax > 0.f ? kmax * (1.0f / 127.0f) : 1.0f;
  const float vscale = vmax > 0.f ? vmax * (1.0f / 127.0f) : 1.0f;

  const float kvv = (float)k[idx];  // k is already (b,h,t,hs)
  kq[idx] = fminf(fmaxf(rintf(kvv / kscale), -127.f), 127.f);

  const int hs = idx & 63;
  const int t  = (idx >> 6) & (Tn - 1);
  const int bh = idx >> 17;
  const float vvv = (float)vt[((size_t)bh * HSn + hs) * Tn + t];
  vq[idx] = fminf(fmaxf(rintf(vvv / vscale), -127.f), 127.f);

  if (idx == 0) { *ks = kscale; *vs = vscale; }
}

// ---------------- Flash attention (WMMA), one wave per (b,h,16-q tile) -----

#define NEG_BIG (-1.0e30f)

__global__ __launch_bounds__(256) void flash_kernel(
    const _Float16* __restrict__ q, const _Float16* __restrict__ k,
    const _Float16* __restrict__ vt, _Float16* __restrict__ y) {
  __shared__ _Float16 plds_all[8][16 * 32];   // per-wave P staging tile
  const int wave = threadIdx.x >> 5;
  const int lane = threadIdx.x & 31;
  const int lm   = lane & 15;
  const int hf   = lane >> 4;
  _Float16* plds = plds_all[wave];

  const int gw = blockIdx.x * 8 + wave;  // 8192 waves total
  const int qt = gw & 127;
  const int bh = gw >> 7;
  const int b  = bh >> 4;
  const int h  = bh & 15;
  const int q0 = qt * 16;

  const _Float16* qp = q  + (size_t)bh * Tn * HSn;
  const _Float16* kp = k  + (size_t)bh * Tn * HSn;
  const _Float16* vp = vt + (size_t)bh * HSn * Tn;

  // Q fragments (row = query, K = hs), loaded once.
  const v16h qf0 = load_a_frag(qp, HSn, q0, 0,  lm, hf);
  const v16h qf1 = load_a_frag(qp, HSn, q0, 32, lm, hf);

  v8f o[4] = {};
  float mrun[8], lrun[8];
#pragma unroll
  for (int r = 0; r < 8; ++r) { mrun[r] = NEG_BIG; lrun[r] = 0.f; }

  const int kend = q0 + 16;  // causal: keys < q0+16
  for (int kb = 0; kb < kend; kb += 32) {
    // ---- S = Q K^T for two 16-key tiles -------------------------------
    v8f s[2];
#pragma unroll
    for (int j = 0; j < 2; ++j) {
      const int key0 = kb + j * 16;
      // B-fragment: contraction = hs; col = key; contiguous in hs.
      const v16h b0 = *(const v16h*)(kp + (size_t)(key0 + lm) * HSn + 0  + hf * 16);
      const v16h b1 = *(const v16h*)(kp + (size_t)(key0 + lm) * HSn + 32 + hf * 16);
      v8f c = {};
      c = wmma16(qf0, b0, c);
      c = wmma16(qf1, b1, c);
      const int keyg = key0 + lm;
#pragma unroll
      for (int r = 0; r < 8; ++r) {
        const int qg = q0 + r + hf * 8;
        s[j][r] = (keyg <= qg) ? c[r] * 0.125f : NEG_BIG;  // 1/sqrt(64)
      }
    }

    // ---- online softmax stats ----------------------------------------
    float mnew[8], alpha[8];
#pragma unroll
    for (int r = 0; r < 8; ++r) {
      const float rm = fmaxf(rmax16(s[0][r]), rmax16(s[1][r]));
      mnew[r]  = fmaxf(mrun[r], rm);
      alpha[r] = expf(mrun[r] - mnew[r]);
    }
#pragma unroll
    for (int j = 0; j < 2; ++j)
#pragma unroll
      for (int r = 0; r < 8; ++r) s[j][r] = expf(s[j][r] - mnew[r]);
#pragma unroll
    for (int r = 0; r < 8; ++r) {
      lrun[r] = lrun[r] * alpha[r] + rsum16(s[0][r]) + rsum16(s[1][r]);
      mrun[r] = mnew[r];
    }
#pragma unroll
    for (int nt = 0; nt < 4; ++nt)
#pragma unroll
      for (int r = 0; r < 8; ++r) o[nt][r] *= alpha[r];

    // ---- P (C-layout f32) -> LDS -> A-fragment f16 --------------------
#pragma unroll
    for (int j = 0; j < 2; ++j)
#pragma unroll
      for (int r = 0; r < 8; ++r)
        plds[(r + hf * 8) * 32 + j * 16 + lm] = (_Float16)s[j][r];
    asm volatile("s_wait_dscnt 0" ::: "memory");

    v16h pf;
    {
      const _Float16* pp = plds + lm * 32 + hf * 16;
      const v8h lo = *(const v8h*)pp;
      const v8h hi = *(const v8h*)(pp + 8);
#pragma unroll
      for (int i = 0; i < 8; ++i) { pf[i] = lo[i]; pf[i + 8] = hi[i]; }
    }

    // ---- O += P * V (contraction = 32 keys, vT contiguous in key) -----
#pragma unroll
    for (int nt = 0; nt < 4; ++nt) {
      const v16h vf = *(const v16h*)(vp + (size_t)(nt * 16 + lm) * Tn + kb + hf * 16);
      o[nt] = wmma16(pf, vf, o[nt]);
    }
  }

  // ---- epilogue: y(B,T,C) f16, heads merged --------------------------
#pragma unroll
  for (int nt = 0; nt < 4; ++nt)
#pragma unroll
    for (int r = 0; r < 8; ++r) {
      const int t = q0 + r + hf * 8;
      const int c = h * HSn + nt * 16 + lm;
      y[((size_t)b * Tn + t) * Cn + c] = (_Float16)(o[nt][r] / lrun[r]);
    }
}

// ---------------------------------------------------------------------------

extern "C" void kernel_launch(void* const* d_in, const int* in_sizes, int n_in,
                              void* d_out, int out_size, void* d_ws, size_t ws_size,
                              hipStream_t stream) {
  const float* x     = (const float*)d_in[0];
  const float* ln1_w = (const float*)d_in[1];
  const float* ln1_b = (const float*)d_in[2];
  const float* W_qkv = (const float*)d_in[3];
  const float* b_qkv = (const float*)d_in[4];
  const float* W_o   = (const float*)d_in[5];
  const float* b_o   = (const float*)d_in[6];
  const float* ln2_w = (const float*)d_in[7];
  const float* ln2_b = (const float*)d_in[8];
  const float* W_fc  = (const float*)d_in[9];
  const float* b_fc  = (const float*)d_in[10];
  const float* W_pr  = (const float*)d_in[11];
  const float* b_pr  = (const float*)d_in[12];

  // ---- workspace layout ----
  char* ws = (char*)d_ws;
  size_t off = 0;
  auto carve = [&](size_t bytes) {
    char* p = ws + off;
    off += (bytes + 255) & ~(size_t)255;
    return p;
  };
  _Float16* h16    = (_Float16*)carve((size_t)Mrows * Cn * 2);
  _Float16* wt_qkv = (_Float16*)carve((size_t)3 * Cn * Cn * 2);
  _Float16* qkv16  = (_Float16*)carve((size_t)Mrows * 3 * Cn * 2);
  _Float16* q16    = (_Float16*)carve((size_t)Mrows * Cn * 2);
  _Float16* k16    = (_Float16*)carve((size_t)Mrows * Cn * 2);
  _Float16* v16t   = (_Float16*)carve((size_t)Mrows * Cn * 2);
  _Float16* y16    = (_Float16*)carve((size_t)Mrows * Cn * 2);
  _Float16* wt_o   = (_Float16*)carve((size_t)Cn * Cn * 2);
  float*    x1     = (float*)   carve((size_t)Mrows * Cn * 4);
  _Float16* h2     = (_Float16*)carve((size_t)Mrows * Cn * 2);
  _Float16* wt_fc  = (_Float16*)carve((size_t)4 * Cn * Cn * 2);
  _Float16* a2     = (_Float16*)carve((size_t)Mrows * 4 * Cn * 2);
  _Float16* wt_pr  = (_Float16*)carve((size_t)4 * Cn * Cn * 2);
  unsigned int* absmax = (unsigned int*)carve(8);

  // ---- output layout: [x | k_quant | k_scale | v_quant | v_scale] ----
  float* out   = (float*)d_out;
  float* kq    = out + (size_t)Mrows * Cn;
  float* ksc   = kq  + (size_t)Mrows * Cn;
  float* vq    = ksc + 1;
  float* vsc   = vq  + (size_t)Mrows * Cn;

  // ---- pipeline ----
  hipMemsetAsync(absmax, 0, 8, stream);

  // weight transposes (f32 KxN -> f16 NxK)
  wtrans_kernel<<<dim3(3 * Cn / 32, Cn / 32), dim3(32, 32), 0, stream>>>(W_qkv, wt_qkv, Cn, 3 * Cn);
  wtrans_kernel<<<dim3(Cn / 32, Cn / 32),     dim3(32, 32), 0, stream>>>(W_o,   wt_o,   Cn, Cn);
  wtrans_kernel<<<dim3(4 * Cn / 32, Cn / 32), dim3(32, 32), 0, stream>>>(W_fc,  wt_fc,  Cn, 4 * Cn);
  wtrans_kernel<<<dim3(Cn / 32, 4 * Cn / 32), dim3(32, 32), 0, stream>>>(W_pr,  wt_pr,  4 * Cn, Cn);

  // LN1 -> h16
  ln_kernel<<<Mrows, 256, 0, stream>>>(x, ln1_w, ln1_b, h16);

  // qkv = h @ W_qkv + b_qkv
  gemm_kernel<<<dim3(Mrows / 256, 3 * Cn / 64), 256, 0, stream>>>(
      h16, wt_qkv, b_qkv, nullptr, qkv16, nullptr, Mrows, 3 * Cn, Cn, 0);

  // split/reshape + absmax
  reshape_qkv_kernel<<<(size_t)Mrows * Cn / 256, 256, 0, stream>>>(
      qkv16, q16, k16, v16t, absmax);

  // int8 KV cache outputs
  quantize_kernel<<<(size_t)Mrows * Cn / 256, 256, 0, stream>>>(
      k16, v16t, absmax, kq, ksc, vq, vsc);

  // flash attention -> y16 (B,T,C)
  flash_kernel<<<Bn * Hn * (Tn / 16) / 8, 256, 0, stream>>>(q16, k16, v16t, y16);

  // x1 = x + y @ W_o + b_o
  gemm_kernel<<<dim3(Mrows / 256, Cn / 64), 256, 0, stream>>>(
      y16, wt_o, b_o, x, nullptr, x1, Mrows, Cn, Cn, 2);

  // LN2 -> h2
  ln_kernel<<<Mrows, 256, 0, stream>>>(x1, ln2_w, ln2_b, h2);

  // a2 = gelu(h2 @ W_fc + b_fc)
  gemm_kernel<<<dim3(Mrows / 256, 4 * Cn / 64), 256, 0, stream>>>(
      h2, wt_fc, b_fc, nullptr, a2, nullptr, Mrows, 4 * Cn, Cn, 1);

  // out = x1 + a2 @ W_pr + b_pr
  gemm_kernel<<<dim3(Mrows / 256, Cn / 64), 256, 0, stream>>>(
      a2, wt_pr, b_pr, x1, nullptr, out, Mrows, Cn, 4 * Cn, 2);

  (void)in_sizes; (void)n_in; (void)out_size; (void)ws_size;
}
